// TARN_44384192036910
// MI455X (gfx1250) — compile-verified, matrix operand order checked by reading, not run on hardware
//
#include <hip/hip_runtime.h>
#include <hip/hip_bf16.h>
#include <math.h>

typedef unsigned short u16;
typedef __attribute__((ext_vector_type(16))) __bf16 v16bf;
typedef __attribute__((ext_vector_type(8)))  float  v8f;

// ---------------------------------------------------------------- helpers ---
__device__ __forceinline__ u16 f2bf(float f) {
  unsigned int u = __float_as_uint(f);
  u += 0x7FFFu + ((u >> 16) & 1u);       // round-to-nearest-even
  return (u16)(u >> 16);
}
__device__ __forceinline__ float sigm(float x) { return 1.0f / (1.0f + expf(-x)); }

// gfx1250 async global->LDS copy (16B per lane), tracked by ASYNCcnt
__device__ __forceinline__ void async_b128(const u16* smem_dst, const u16* gsrc) {
  unsigned lds = (unsigned)(unsigned long long)smem_dst;       // LDS byte offset
  unsigned long long g = (unsigned long long)gsrc;
  asm volatile("global_load_async_to_lds_b128 %0, %1, off"
               :: "v"(lds), "v"(g) : "memory");
}
__device__ __forceinline__ void wait_async0() {
  asm volatile("s_wait_asynccnt 0x0" ::: "memory");
}
__device__ __forceinline__ void wait_async2() {
  asm volatile("s_wait_asynccnt 0x2" ::: "memory");
}

// K index pattern for 16-bit WMMA fragments (ISA 7.12.2):
// lanes 0-15: elems 0..7 -> K 0..7,  elems 8..15 -> K 16..23
// lanes16-31: elems 0..7 -> K 8..15, elems 8..15 -> K 24..31
__device__ __forceinline__ int kpat(int i, int hi) {
  return (i < 8) ? (hi * 8 + i) : (8 + hi * 8 + i);
}
// A fragment from [M][K] storage (row-major per m), ld in elements
__device__ __forceinline__ v16bf frag_a_mk(const u16* p, int ld) {
  const int lane = threadIdx.x & 31, m = lane & 15, hi = lane >> 4;
  v16bf a;
#pragma unroll
  for (int i = 0; i < 16; ++i) a[i] = __builtin_bit_cast(__bf16, p[m * ld + kpat(i, hi)]);
  return a;
}
// A fragment from [K][M] storage (transposed access)
__device__ __forceinline__ v16bf frag_a_km(const u16* p, int ld) {
  const int lane = threadIdx.x & 31, m = lane & 15, hi = lane >> 4;
  v16bf a;
#pragma unroll
  for (int i = 0; i < 16; ++i) a[i] = __builtin_bit_cast(__bf16, p[kpat(i, hi) * ld + m]);
  return a;
}
// B fragment from [N][K] storage
__device__ __forceinline__ v16bf frag_b_nk(const u16* p, int ld) {
  const int lane = threadIdx.x & 31, n = lane & 15, hi = lane >> 4;
  v16bf b;
#pragma unroll
  for (int i = 0; i < 16; ++i) b[i] = __builtin_bit_cast(__bf16, p[n * ld + kpat(i, hi)]);
  return b;
}
// B fragment from [K][N] storage
__device__ __forceinline__ v16bf frag_b_kn(const u16* p, int ld) {
  const int lane = threadIdx.x & 31, n = lane & 15, hi = lane >> 4;
  v16bf b;
#pragma unroll
  for (int i = 0; i < 16; ++i) b[i] = __builtin_bit_cast(__bf16, p[kpat(i, hi) * ld + n]);
  return b;
}
__device__ __forceinline__ v8f wmma_bf16(v16bf a, v16bf b, v8f c) {
  return __builtin_amdgcn_wmma_f32_16x16x32_bf16(false, a, false, b, (short)0, c, false, false);
}

// -------------------------------------------------------- workspace layout ---
static constexpr size_t SZ_W1     = 1536ull * 4096 * 2;   // bf16 [Wih_f;Wih_b]
static constexpr size_t SZ_WHH    = 768ull * 256 * 2;
static constexpr size_t SZ_WATTN  = 512ull * 512 * 2;
static constexpr size_t SZ_XP     = 12288ull * 1536 * 4;  // fp32 input projections
static constexpr size_t SZ_F      = 12288ull * 512 * 4;   // fp32 BiGRU outputs (S then Q)
static constexpr size_t SZ_FBF    = 12288ull * 512 * 2;
static constexpr size_t SZ_KEYS   = 10240ull * 512 * 2;
static constexpr size_t SZ_A      = 160ull * 4096 * 4;
static constexpr size_t SZ_ABF    = 160ull * 4096 * 2;
static constexpr size_t SZ_HC     = 10240ull * 512 * 4;

static constexpr size_t OFF_W1     = 0;
static constexpr size_t OFF_WHHF   = OFF_W1 + SZ_W1;
static constexpr size_t OFF_WHHB   = OFF_WHHF + SZ_WHH;
static constexpr size_t OFF_WATTN  = OFF_WHHB + SZ_WHH;
static constexpr size_t OFF_DMLWHH = OFF_WATTN + SZ_WATTN;
static constexpr size_t OFF_XP     = OFF_DMLWHH + SZ_WHH;
static constexpr size_t OFF_F      = OFF_XP + SZ_XP;
static constexpr size_t OFF_FBF    = OFF_F + SZ_F;
static constexpr size_t OFF_KEYS   = OFF_FBF + SZ_FBF;
static constexpr size_t OFF_A      = OFF_KEYS + SZ_KEYS;
static constexpr size_t OFF_ABF    = OFF_A + SZ_A;
static constexpr size_t OFF_HC     = OFF_ABF + SZ_ABF;
static constexpr size_t OFF_DIFF   = OFF_HC + SZ_HC;      // + 10240*2*4 => ~163 MB total

// ------------------------------------------------- 0: weight bf16 convert ---
__global__ void __launch_bounds__(256) k_convert(
    const float* __restrict__ WihF, const float* __restrict__ WihB,
    const float* __restrict__ WhhF, const float* __restrict__ WhhB,
    const float* __restrict__ Wattn, const float* __restrict__ dWhh,
    u16* __restrict__ w1, u16* __restrict__ whhf, u16* __restrict__ whhb,
    u16* __restrict__ wattn, u16* __restrict__ dwhh) {
  const int n_w1 = 1536 * 4096, n_whh = 768 * 256, n_attn = 512 * 512;
  const int total = n_w1 + 3 * n_whh + n_attn;
  for (int i = blockIdx.x * 256 + threadIdx.x; i < total; i += gridDim.x * 256) {
    if (i < n_w1) {
      int g = i >> 12, d = i & 4095;
      w1[i] = f2bf(g < 768 ? WihF[g * 4096 + d] : WihB[(g - 768) * 4096 + d]);
    } else {
      int j = i - n_w1;
      if (j < n_whh) whhf[j] = f2bf(WhhF[j]);
      else if (j < 2 * n_whh) whhb[j - n_whh] = f2bf(WhhB[j - n_whh]);
      else if (j < 2 * n_whh + n_attn) wattn[j - 2 * n_whh] = f2bf(Wattn[j - 2 * n_whh]);
      else dwhh[j - 2 * n_whh - n_attn] = f2bf(dWhh[j - 2 * n_whh - n_attn]);
    }
  }
}

// ------- 1: xp = X @ [Wih_f;Wih_b]^T + bias (WMMA bf16, async B pipeline) ---
// X rows: 0..10239 = feats_S (K*B*TS), 10240..12287 = feats_Q
__global__ void __launch_bounds__(256) k_gemm_xp(
    const float* __restrict__ fS, const float* __restrict__ fQ,
    const u16* __restrict__ W1, const float* __restrict__ bihF,
    const float* __restrict__ bihB, float* __restrict__ xp) {
  __shared__ u16 As[64][80];        // 64 rows x 64 K, bf16 (converted from fp32)
  __shared__ u16 Bs[2][64][80];     // double-buffered weight tile, async-filled
  const int m0 = blockIdx.x * 64, n0 = blockIdx.y * 64;
  const int tid = threadIdx.x, wave = tid >> 5, lane = tid & 31;
  const int hi = lane >> 4, nl = lane & 15;
  const int t0 = wave * 2, mt = t0 >> 2, nt0 = t0 & 3, nt1 = nt0 + 1;
  const int lr = tid >> 2, lc = (tid & 3) * 16;     // 16 elems per thread per row
  const int R = m0 + lr;
  const float* srcRow = (R < 10240) ? (fS + (size_t)R * 4096)
                                    : (fQ + (size_t)(R - 10240) * 4096);
  const u16* wRow = W1 + (size_t)(n0 + lr) * 4096;
  // prologue: async-stage weight chunk 0
  async_b128(&Bs[0][lr][lc],     wRow + lc);
  async_b128(&Bs[0][lr][lc + 8], wRow + lc + 8);
  v8f acc0 = {}, acc1 = {};
  for (int it = 0; it < 64; ++it) {
    const int k0 = it * 64;
    const int cur = it & 1, nxt = cur ^ 1;
    // A stage: fp32 -> bf16 into LDS
#pragma unroll
    for (int j = 0; j < 4; ++j) {
      float4 f = *(const float4*)(srcRow + k0 + lc + j * 4);
      As[lr][lc + j * 4 + 0] = f2bf(f.x);
      As[lr][lc + j * 4 + 1] = f2bf(f.y);
      As[lr][lc + j * 4 + 2] = f2bf(f.z);
      As[lr][lc + j * 4 + 3] = f2bf(f.w);
    }
    // prefetch next weight chunk while this one is consumed
    if (it + 1 < 64) {
      async_b128(&Bs[nxt][lr][lc],     wRow + k0 + 64 + lc);
      async_b128(&Bs[nxt][lr][lc + 8], wRow + k0 + 64 + lc + 8);
      wait_async2();                  // <=2 outstanding => current buffer landed
    } else {
      wait_async0();
    }
    __syncthreads();
#pragma unroll
    for (int ks = 0; ks < 2; ++ks) {
      v16bf a  = frag_a_mk(&As[mt * 16][ks * 32], 80);
      v16bf b0 = frag_b_nk(&Bs[cur][nt0 * 16][ks * 32], 80);
      v16bf b1 = frag_b_nk(&Bs[cur][nt1 * 16][ks * 32], 80);
      acc0 = wmma_bf16(a, b0, acc0);
      acc1 = wmma_bf16(a, b1, acc1);
    }
    __syncthreads();
  }
  const int mb = m0 + mt * 16 + 8 * hi;
  {
    const int g = n0 + nt0 * 16 + nl;
    const float bias = (g < 768) ? bihF[g] : bihB[g - 768];
#pragma unroll
    for (int v = 0; v < 8; ++v) xp[(size_t)(mb + v) * 1536 + g] = acc0[v] + bias;
  }
  {
    const int g = n0 + nt1 * 16 + nl;
    const float bias = (g < 768) ? bihF[g] : bihB[g - 768];
#pragma unroll
    for (int v = 0; v < 8; ++v) xp[(size_t)(mb + v) * 1536 + g] = acc1[v] + bias;
  }
}

// ----------------- 2: GRU recurrence (both directions), h@Whh^T via WMMA ----
// grid (12, 2): 16 sequences per block (192 seqs of length 64), y=direction
__global__ void __launch_bounds__(256) k_gru_main(
    const float* __restrict__ xp, const u16* __restrict__ WhhF,
    const u16* __restrict__ WhhB, const float* __restrict__ bhhF,
    const float* __restrict__ bhhB, float* __restrict__ F, u16* __restrict__ Fbf) {
  __shared__ u16  h_bf[16][264];
  __shared__ float hp[16][772];
  __shared__ float h_f[16][256];
  const int tid = threadIdx.x;
  const int s0  = blockIdx.x * 16;
  const int dir = blockIdx.y;
  const u16*   Whh = dir ? WhhB : WhhF;
  const float* bhh = dir ? bhhB : bhhF;
  const int goff = dir * 768;
  for (int i = tid; i < 16 * 264; i += 256) ((u16*)h_bf)[i] = 0;
  for (int i = tid; i < 16 * 256; i += 256) ((float*)h_f)[i] = 0.0f;
  __syncthreads();
  const int wave = tid >> 5, lane = tid & 31, hi = lane >> 4, nl = lane & 15;
  for (int t = 0; t < 64; ++t) {
    for (int jj = 0; jj < 6; ++jj) {              // 48 N-tiles / 8 waves
      const int j = wave * 6 + jj;
      const u16* bb = Whh + (size_t)(j * 16) * 256;
      __builtin_prefetch(bb, 0, 3);               // L2-resident weight rows
      v8f acc = {};
#pragma unroll
      for (int kt = 0; kt < 8; ++kt) {
        v16bf a = frag_a_mk(&h_bf[0][kt * 32], 264);
        v16bf b = frag_b_nk(bb + kt * 32, 256);
        acc = wmma_bf16(a, b, acc);
      }
#pragma unroll
      for (int v = 0; v < 8; ++v) hp[8 * hi + v][j * 16 + nl] = acc[v];
    }
    __syncthreads();
    const int tx = dir ? (63 - t) : t;            // bwd consumes reversed input
    for (int idx = tid; idx < 4096; idx += 256) {
      const int m = idx >> 8, hh = idx & 255;
      const int sq = s0 + m;
      const float* xr = xp + (size_t)(sq * 64 + tx) * 1536 + goff;
      const float r = sigm(xr[hh]       + hp[m][hh]       + bhh[hh]);
      const float z = sigm(xr[256 + hh] + hp[m][256 + hh] + bhh[256 + hh]);
      const float n = tanhf(xr[512 + hh] + r * (hp[m][512 + hh] + bhh[512 + hh]));
      const float hn = (1.0f - z) * n + z * h_f[m][hh];
      h_f[m][hh] = hn;
      h_bf[m][hh] = f2bf(hn);
      const size_t orow = (size_t)(sq * 64 + tx) * 512 + dir * 256 + hh; // bwd un-reversed
      F[orow] = hn;
      Fbf[orow] = f2bf(hn);
    }
    __syncthreads();
  }
}

// ------- 3: keys = F_S @ W_attn^T + attn_bias (bf16 out, async staging) -----
__global__ void __launch_bounds__(256) k_keys(
    const u16* __restrict__ Fbf, const u16* __restrict__ Wattn,
    const float* __restrict__ abias, u16* __restrict__ keys) {
  __shared__ u16 As[64][48];
  __shared__ u16 Bs[64][48];
  const int m0 = blockIdx.x * 64, n0 = blockIdx.y * 64;
  const int tid = threadIdx.x, wave = tid >> 5, lane = tid & 31;
  const int hi = lane >> 4, nl = lane & 15;
  const int t0 = wave * 2, mt = t0 >> 2, nt0 = t0 & 3, nt1 = nt0 + 1;
  const int lr = tid >> 2, lc = (tid & 3) * 8;
  v8f acc0 = {}, acc1 = {};
  for (int k0 = 0; k0 < 512; k0 += 32) {
    async_b128(&As[lr][lc], Fbf + (size_t)(m0 + lr) * 512 + k0 + lc);
    async_b128(&Bs[lr][lc], Wattn + (size_t)(n0 + lr) * 512 + k0 + lc);
    wait_async0();
    __syncthreads();
    v16bf a  = frag_a_mk(&As[mt * 16][0], 48);
    v16bf b0 = frag_b_nk(&Bs[nt0 * 16][0], 48);
    v16bf b1 = frag_b_nk(&Bs[nt1 * 16][0], 48);
    acc0 = wmma_bf16(a, b0, acc0);
    acc1 = wmma_bf16(a, b1, acc1);
    __syncthreads();
  }
  const int mb = m0 + mt * 16 + 8 * hi;
  {
    const int g = n0 + nt0 * 16 + nl; const float bias = abias[g];
#pragma unroll
    for (int v = 0; v < 8; ++v) keys[(size_t)(mb + v) * 512 + g] = f2bf(acc0[v] + bias);
  }
  {
    const int g = n0 + nt1 * 16 + nl; const float bias = abias[g];
#pragma unroll
    for (int v = 0; v < 8; ++v) keys[(size_t)(mb + v) * 512 + g] = f2bf(acc1[v] + bias);
  }
}

// ------ 4: per-(k,b) scores = keys @ F_Q^T, masked softmax, s-mask, A out ---
__global__ void __launch_bounds__(256) k_scores(
    const u16* __restrict__ keys, const u16* __restrict__ Fbf,
    const int* __restrict__ sln, const int* __restrict__ qln,
    float* __restrict__ A, u16* __restrict__ Abf, float* __restrict__ outA) {
  __shared__ u16  Ks[64][48];
  __shared__ u16  Qs[64][48];
  __shared__ float Sc[64][68];
  const int sqs = blockIdx.x;          // kk*32 + b
  const int b = sqs & 31;
  const int tid = threadIdx.x, wave = tid >> 5, lane = tid & 31;
  const int hi = lane >> 4, nl = lane & 15;
  const int t0 = wave * 2, mt = t0 >> 2, nt0 = t0 & 3, nt1 = nt0 + 1;
  const int lr = tid >> 2, lc = (tid & 3) * 8;
  const size_t krow = (size_t)sqs * 64;
  const size_t qrow = (size_t)(160 + b) * 64;
  v8f acc0 = {}, acc1 = {};
  for (int k0 = 0; k0 < 512; k0 += 32) {
    *(uint4*)&Ks[lr][lc] = *(const uint4*)(keys + (krow + lr) * 512 + k0 + lc);
    *(uint4*)&Qs[lr][lc] = *(const uint4*)(Fbf + (qrow + lr) * 512 + k0 + lc);
    __syncthreads();
    v16bf a  = frag_a_mk(&Ks[mt * 16][0], 48);
    v16bf b0 = frag_b_nk(&Qs[nt0 * 16][0], 48);
    v16bf b1 = frag_b_nk(&Qs[nt1 * 16][0], 48);
    acc0 = wmma_bf16(a, b0, acc0);
    acc1 = wmma_bf16(a, b1, acc1);
    __syncthreads();
  }
#pragma unroll
  for (int v = 0; v < 8; ++v) {
    Sc[mt * 16 + 8 * hi + v][nt0 * 16 + nl] = acc0[v];
    Sc[mt * 16 + 8 * hi + v][nt1 * 16 + nl] = acc1[v];
  }
  __syncthreads();
  if (tid < 64) {
    const int s = tid;
    const int ql = qln[b], sl = sln[sqs];
    const float sm = (s < sl) ? 1.0f : 0.0f;
    float mx = -1e30f;
    for (int q = 0; q < 64; ++q) {
      float v = (q < ql) ? Sc[s][q] : -1e30f;
      Sc[s][q] = v; mx = fmaxf(mx, v);
    }
    float sum = 0.0f;
    for (int q = 0; q < 64; ++q) { float e = expf(Sc[s][q] - mx); Sc[s][q] = e; sum += e; }
    const float inv = sm / sum;
    for (int q = 0; q < 64; ++q) {
      float a = Sc[s][q] * inv;
      A[(size_t)sqs * 4096 + s * 64 + q] = a;
      Abf[(size_t)sqs * 4096 + s * 64 + q] = f2bf(a);
      if (sqs == 159) outA[s * 64 + q] = a;     // last (class,sample) pair -> output
    }
  }
}

// ------------------------ 5: Hc = A^T @ F_S per (k,b)  (K-major fragments) ---
__global__ void __launch_bounds__(256) k_hc(
    const u16* __restrict__ Abf, const u16* __restrict__ Fbf, float* __restrict__ Hc) {
  __shared__ u16 As[64][80];   // [s][q]
  __shared__ u16 Bs[32][80];   // [s][h-chunk]
  const int sqs = blockIdx.x;
  const int h0 = blockIdx.y * 64;
  const int tid = threadIdx.x, wave = tid >> 5, lane = tid & 31;
  const int hi = lane >> 4, nl = lane & 15;
  const int t0 = wave * 2, qt = t0 >> 2, ht0 = t0 & 3, ht1 = ht0 + 1;
  {
    const int r = tid >> 2, cb = (tid & 3) * 16;
    *(uint4*)&As[r][cb]     = *(const uint4*)(Abf + (size_t)sqs * 4096 + r * 64 + cb);
    *(uint4*)&As[r][cb + 8] = *(const uint4*)(Abf + (size_t)sqs * 4096 + r * 64 + cb + 8);
  }
  const int lr = tid >> 3, lc = (tid & 7) * 8;
  v8f acc0 = {}, acc1 = {};
  for (int s0 = 0; s0 < 64; s0 += 32) {
    *(uint4*)&Bs[lr][lc] =
        *(const uint4*)(Fbf + (size_t)(sqs * 64 + s0 + lr) * 512 + h0 + lc);
    __syncthreads();
    v16bf a  = frag_a_km(&As[s0][qt * 16], 80);
    v16bf b0 = frag_b_kn(&Bs[0][ht0 * 16], 80);
    v16bf b1 = frag_b_kn(&Bs[0][ht1 * 16], 80);
    acc0 = wmma_bf16(a, b0, acc0);
    acc1 = wmma_bf16(a, b1, acc1);
    __syncthreads();
  }
  const int qb = qt * 16 + 8 * hi;
#pragma unroll
  for (int v = 0; v < 8; ++v) {
    Hc[(size_t)(sqs * 64 + qb + v) * 512 + h0 + ht0 * 16 + nl] = acc0[v];
    Hc[(size_t)(sqs * 64 + qb + v) * 512 + h0 + ht1 * 16 + nl] = acc1[v];
  }
}

// --------------------- 6: cosine / euclidean distances, wave per (k,b,q) ----
__global__ void __launch_bounds__(256) k_diff(
    const float* __restrict__ F, const float* __restrict__ Hc,
    const int* __restrict__ qln, float* __restrict__ diff) {
  const int item = blockIdx.x * 8 + (threadIdx.x >> 5);
  const int lane = threadIdx.x & 31;
  if (item >= 10240) return;
  const int b = (item >> 6) & 31, q = item & 63;
  const float* hq = F + ((size_t)(160 + b) * 64 + q) * 512;
  const float* hc = Hc + (size_t)item * 512;
  float dot = 0, nq = 0, nh = 0, dd = 0;
  for (int c = lane; c < 512; c += 32) {
    float a = hq[c], h = hc[c];
    dot += a * h; nq += a * a; nh += h * h;
    float df = a - h; dd += df * df;
  }
  for (int off = 16; off > 0; off >>= 1) {
    dot += __shfl_xor(dot, off); nq += __shfl_xor(nq, off);
    nh  += __shfl_xor(nh, off);  dd += __shfl_xor(dd, off);
  }
  if (lane == 0) {
    const float m = (q < qln[b]) ? 1.0f : 0.0f;
    diff[item * 2 + 0] = (dot / fmaxf(sqrtf(nq) * sqrtf(nh), 1e-6f)) * m;
    diff[item * 2 + 1] = sqrtf(dd) * m;
  }
}

// --------- 7: dml GRU over diff (input dim 2) + fc sigmoid -> q_kc output ---
__global__ void __launch_bounds__(256) k_dml(
    const float* __restrict__ diff, const u16* __restrict__ Whh,
    const float* __restrict__ Wih, const float* __restrict__ bih,
    const float* __restrict__ bhh, const float* __restrict__ fcW,
    const float* __restrict__ fcb, float* __restrict__ outq) {
  __shared__ u16  h_bf[16][264];
  __shared__ float hp[16][772];
  __shared__ float h_f[16][256];
  const int tid = threadIdx.x;
  const int s0 = blockIdx.x * 16;
  for (int i = tid; i < 16 * 264; i += 256) ((u16*)h_bf)[i] = 0;
  for (int i = tid; i < 16 * 256; i += 256) ((float*)h_f)[i] = 0.0f;
  __syncthreads();
  const int wave = tid >> 5, lane = tid & 31, hi = lane >> 4, nl = lane & 15;
  for (int t = 0; t < 64; ++t) {
    for (int jj = 0; jj < 6; ++jj) {
      const int j = wave * 6 + jj;
      const u16* bb = Whh + (size_t)(j * 16) * 256;
      __builtin_prefetch(bb, 0, 3);
      v8f acc = {};
#pragma unroll
      for (int kt = 0; kt < 8; ++kt) {
        v16bf a = frag_a_mk(&h_bf[0][kt * 32], 264);
        v16bf b = frag_b_nk(bb + kt * 32, 256);
        acc = wmma_bf16(a, b, acc);
      }
#pragma unroll
      for (int v = 0; v < 8; ++v) hp[8 * hi + v][j * 16 + nl] = acc[v];
    }
    __syncthreads();
    for (int idx = tid; idx < 4096; idx += 256) {
      const int m = idx >> 8, hh = idx & 255;
      const int sq = s0 + m;
      const float d0 = diff[(size_t)(sq * 64 + t) * 2];
      const float d1 = diff[(size_t)(sq * 64 + t) * 2 + 1];
      const float xr = Wih[hh * 2] * d0 + Wih[hh * 2 + 1] * d1 + bih[hh];
      const float xz = Wih[(256 + hh) * 2] * d0 + Wih[(256 + hh) * 2 + 1] * d1 + bih[256 + hh];
      const float xn = Wih[(512 + hh) * 2] * d0 + Wih[(512 + hh) * 2 + 1] * d1 + bih[512 + hh];
      const float r = sigm(xr + hp[m][hh] + bhh[hh]);
      const float z = sigm(xz + hp[m][256 + hh] + bhh[256 + hh]);
      const float n = tanhf(xn + r * (hp[m][512 + hh] + bhh[512 + hh]));
      const float hn = (1.0f - z) * n + z * h_f[m][hh];
      h_f[m][hh] = hn;
      h_bf[m][hh] = f2bf(hn);
    }
    __syncthreads();
  }
  if (tid < 16) {
    float acc = fcb[0];
    for (int c = 0; c < 256; ++c) acc += h_f[tid][c] * fcW[c];
    outq[4096 + s0 + tid] = 1.0f / (1.0f + expf(-acc));
  }
}

// ---------------------------------------------------------------- launcher ---
extern "C" void kernel_launch(void* const* d_in, const int* in_sizes, int n_in,
                              void* d_out, int out_size, void* d_ws, size_t ws_size,
                              hipStream_t stream) {
  (void)in_sizes; (void)n_in; (void)out_size; (void)ws_size;
  const float* fS    = (const float*)d_in[0];
  const float* fQ    = (const float*)d_in[1];
  const float* WihF  = (const float*)d_in[2];
  const float* WhhF  = (const float*)d_in[3];
  const float* bihF  = (const float*)d_in[4];
  const float* bhhF  = (const float*)d_in[5];
  const float* WihB  = (const float*)d_in[6];
  const float* WhhB  = (const float*)d_in[7];
  const float* bihB  = (const float*)d_in[8];
  const float* bhhB  = (const float*)d_in[9];
  const float* Wattn = (const float*)d_in[10];
  const float* abias = (const float*)d_in[11];
  const float* dWih  = (const float*)d_in[12];
  const float* dWhh  = (const float*)d_in[13];
  const float* dbih  = (const float*)d_in[14];
  const float* dbhh  = (const float*)d_in[15];
  const float* fcW   = (const float*)d_in[16];
  const float* fcb   = (const float*)d_in[17];
  const int*   sln   = (const int*)d_in[18];   // [K,B]
  const int*   qln   = (const int*)d_in[19];   // [B]
  float* out = (float*)d_out;

  char* ws = (char*)d_ws;
  u16*   W1_bf   = (u16*)(ws + OFF_W1);
  u16*   WhhF_bf = (u16*)(ws + OFF_WHHF);
  u16*   WhhB_bf = (u16*)(ws + OFF_WHHB);
  u16*   Wat_bf  = (u16*)(ws + OFF_WATTN);
  u16*   dWhh_bf = (u16*)(ws + OFF_DMLWHH);
  float* xp      = (float*)(ws + OFF_XP);
  float* F       = (float*)(ws + OFF_F);
  u16*   Fbf     = (u16*)(ws + OFF_FBF);
  u16*   keys    = (u16*)(ws + OFF_KEYS);
  float* A       = (float*)(ws + OFF_A);
  u16*   Abf     = (u16*)(ws + OFF_ABF);
  float* Hc      = (float*)(ws + OFF_HC);
  float* diff    = (float*)(ws + OFF_DIFF);

  k_convert<<<2048, 256, 0, stream>>>(WihF, WihB, WhhF, WhhB, Wattn, dWhh,
                                      W1_bf, WhhF_bf, WhhB_bf, Wat_bf, dWhh_bf);
  k_gemm_xp<<<dim3(192, 24), 256, 0, stream>>>(fS, fQ, W1_bf, bihF, bihB, xp);
  k_gru_main<<<dim3(12, 2), 256, 0, stream>>>(xp, WhhF_bf, WhhB_bf, bhhF, bhhB, F, Fbf);
  k_keys<<<dim3(160, 8), 256, 0, stream>>>(Fbf, Wat_bf, abias, keys);
  k_scores<<<160, 256, 0, stream>>>(keys, Fbf, sln, qln, A, Abf, out);
  k_hc<<<dim3(160, 8), 256, 0, stream>>>(Abf, Fbf, Hc);
  k_diff<<<1280, 256, 0, stream>>>(F, Hc, qln, diff);
  k_dml<<<10, 256, 0, stream>>>(diff, dWhh_bf, dWih, dbih, dbhh, fcW, fcb, out);
}
